// DecoderME_19774029431202
// MI455X (gfx1250) — matrix operand order repaired and checked by reading, not compiled, not run on hardware
//
#include <hip/hip_runtime.h>

// ---------------------------------------------------------------------------
// DecoderME on gfx1250 — all GEMM-like work on v_wmma_f32_16x16x32_f16.
// One wave (32 lanes, wave32) computes a 16x16 f32 C tile.
// Fragment layouts per CDNA5 ISA 7.12.2:
//   A (16x32 f16): lane l -> row M=l&15; half=(l>>4): elems 0..7 = K[8h..8h+7],
//                  elems 8..15 = K[16+8h..16+8h+7]
//   B (32x16 f16): lane l -> col N=l&15; elems 0..7 = K[8h..8h+7], 8..15 = K[16+8h..16+8h+7]
//   C/D (16x16 f32): elem r -> M = r + 8*half, N = l&15
// All loads are UNCONDITIONAL (addresses clamped, values masked) so the
// compiler emits batched load clauses instead of per-element exec branches.
// ---------------------------------------------------------------------------

typedef __attribute__((ext_vector_type(16))) _Float16 v16h;
typedef __attribute__((ext_vector_type(8)))  float    v8f;

#define WMMA_F16(a, b, c) \
    __builtin_amdgcn_wmma_f32_16x16x32_f16(false, (a), false, (b), (short)0, (c), false, false)

// A fragment from fp32 row pointer (already offset to k0). MASKED multiplies
// by a 0/1 float (conv boundary). KMAX<32 => statically zero-padded upper half.
template<int KMAX, bool MASKED>
__device__ __forceinline__ v16h load_a_frag(const float* __restrict__ rowp,
                                            int lane, float mask) {
    int kb = (lane >> 4) << 3;   // 0 or 8
    v16h a;
    if constexpr (KMAX >= 32) {
        float4 x0 = *(const float4*)(rowp + kb);
        float4 x1 = *(const float4*)(rowp + kb + 4);
        float4 x2 = *(const float4*)(rowp + kb + 16);
        float4 x3 = *(const float4*)(rowp + kb + 20);
        float v[16] = {x0.x, x0.y, x0.z, x0.w, x1.x, x1.y, x1.z, x1.w,
                       x2.x, x2.y, x2.z, x2.w, x3.x, x3.y, x3.z, x3.w};
#pragma unroll
        for (int i = 0; i < 16; ++i)
            a[i] = (_Float16)(MASKED ? v[i] * mask : v[i]);
    } else {                      // KMAX == 16: K = kb..kb+7 valid, rest zero
        float4 x0 = *(const float4*)(rowp + kb);
        float4 x1 = *(const float4*)(rowp + kb + 4);
        float v[8] = {x0.x, x0.y, x0.z, x0.w, x1.x, x1.y, x1.z, x1.w};
#pragma unroll
        for (int i = 0; i < 8; ++i) {
            a[i]     = (_Float16)(MASKED ? v[i] * mask : v[i]);
            a[8 + i] = (_Float16)0.f;
        }
    }
    return a;
}

// B fragment: Bk0 points at row k0 of a (K x LDB) fp32 matrix. LDB is a
// compile-time constant so all 16 loads are immediate offsets off one pointer.
template<int KMAX, int LDB>
__device__ __forceinline__ v16h load_b_frag(const float* __restrict__ Bk0,
                                            int col0, int lane) {
    int n = col0 + (lane & 15);
    v16h b;
    if constexpr (KMAX >= 32) {
        const float* p = Bk0 + (size_t)((lane >> 4) << 3) * LDB + n;
#pragma unroll
        for (int i = 0; i < 16; ++i)
            b[i] = (_Float16)p[(i < 8 ? i : i + 8) * LDB];
    } else {                      // KMAX == 16: K = 8h..8h+7 valid, rest zero
        const float* p = Bk0 + (size_t)((lane >> 4) << 3) * LDB + n;
#pragma unroll
        for (int i = 0; i < 8; ++i) {
            b[i]     = (_Float16)p[i * LDB];
            b[8 + i] = (_Float16)0.f;
        }
    }
    return b;
}

// ---------------------------------------------------------------------------
// Dense GEMM + bias + ReLU (FC layers). grid = (M/16, N/16), block = 32.
// ---------------------------------------------------------------------------
template<int K, int N>
__global__ void gemm_bias_relu(const float* __restrict__ A,
                               const float* __restrict__ Bm,
                               const float* __restrict__ bias,
                               float* __restrict__ C) {
    int lane = threadIdx.x;
    int row0 = blockIdx.x * 16;
    int col0 = blockIdx.y * 16;
    const float* arow = A + (size_t)(row0 + (lane & 15)) * K;
    v8f acc = {};
#pragma unroll
    for (int k0 = 0; k0 < K; k0 += 32) {
        v16h a = load_a_frag<K, false>(arow + k0, lane, 1.f);
        v16h b = load_b_frag<K, N>(Bm + (size_t)k0 * N, col0, lane);
        acc = WMMA_F16(a, b, acc);
    }
    int n  = col0 + (lane & 15);
    float bv = bias[n];
    int m0 = row0 + 8 * (lane >> 4);
#pragma unroll
    for (int r = 0; r < 8; ++r)
        C[(size_t)(m0 + r) * N + n] = fmaxf(acc[r] + bv, 0.f);
}

// ---------------------------------------------------------------------------
// up2: Y[b, 2y+p, 2x+q, o] = sum_c X[b,y,x,c] * W[p,q,c,o]
// One A fragment reused across 4 B matrices. grid = (B*h*w/16, CO/16).
// ---------------------------------------------------------------------------
template<int CI, int CO>
__global__ void up2_wmma(const float* __restrict__ X,
                         const float* __restrict__ W,
                         float* __restrict__ Y, int lh) {
    int lane = threadIdx.x;
    int row0 = blockIdx.x * 16;
    int col0 = blockIdx.y * 16;
    const float* arow = X + (size_t)(row0 + (lane & 15)) * CI;

    v8f acc[4] = {};
#pragma unroll 1
    for (int k0 = 0; k0 < CI; k0 += 32) {
        v16h a = load_a_frag<CI, false>(arow + k0, lane, 1.f);
#pragma unroll
        for (int pq = 0; pq < 4; ++pq) {
            v16h b = load_b_frag<CI, CO>(W + (size_t)pq * CI * CO + (size_t)k0 * CO,
                                         col0, lane);
            acc[pq] = WMMA_F16(a, b, acc[pq]);
        }
    }

    int n      = col0 + (lane & 15);
    int half   = lane >> 4;
    int hwMask = (1 << (2 * lh)) - 1;
    int wMask  = (1 << lh) - 1;
#pragma unroll
    for (int pq = 0; pq < 4; ++pq) {
        int p = pq >> 1, q = pq & 1;
#pragma unroll
        for (int r = 0; r < 8; ++r) {
            int rr  = row0 + r + 8 * half;     // flattened input row (b,y,x)
            int b   = rr >> (2 * lh);
            int rem = rr & hwMask;
            int y   = rem >> lh;
            int x   = rem & wMask;
            size_t o = ((((size_t)b * (2 << lh) + 2 * y + p) * (2 << lh)) + 2 * x + q)
                       * (size_t)CO + n;
            Y[o] = acc[pq][r];
        }
    }
}

// ---------------------------------------------------------------------------
// 3x3 SAME conv: 9-tap accumulated GEMM over K=CI. grid = (B*H*W/16, CO/16).
// Boundary handled by address clamp + value mask (branch-free).
// ---------------------------------------------------------------------------
template<int CI, int CO>
__global__ void conv3_wmma(const float* __restrict__ X,
                           const float* __restrict__ W,
                           float* __restrict__ Y, int H, int Wd) {
    int lane = threadIdx.x;
    int row0 = blockIdx.x * 16;
    int col0 = blockIdx.y * 16;

    int m   = row0 + (lane & 15);
    int hw  = H * Wd;
    int b   = m / hw;
    int rem = m - b * hw;
    int y   = rem / Wd;
    int x   = rem - y * Wd;

    v8f acc = {};
#pragma unroll
    for (int dy = 0; dy < 3; ++dy) {
#pragma unroll
        for (int dx = 0; dx < 3; ++dx) {
            int ys = y + dy - 1;
            int xs = x + dx - 1;
            bool valid = ((unsigned)ys < (unsigned)H) & ((unsigned)xs < (unsigned)Wd);
            float mask = valid ? 1.f : 0.f;
            int ysc = ys < 0 ? 0 : (ys >= H  ? H  - 1 : ys);
            int xsc = xs < 0 ? 0 : (xs >= Wd ? Wd - 1 : xs);
            const float* xp = X + ((size_t)(b * H + ysc) * Wd + xsc) * CI;
            const float* wt = W + (size_t)(dy * 3 + dx) * CI * CO;
            constexpr int KS = (CI + 31) / 32;
#pragma unroll 1
            for (int ks = 0; ks < KS; ++ks) {
                v16h a  = load_a_frag<CI, true>(xp + ks * 32, lane, mask);
                v16h bf = load_b_frag<CI, CO>(wt + (size_t)ks * 32 * CO, col0, lane);
                acc = WMMA_F16(a, bf, acc);
            }
        }
    }

    int n  = col0 + (lane & 15);
    int m0 = row0 + 8 * (lane >> 4);
#pragma unroll
    for (int r = 0; r < 8; ++r)
        Y[(size_t)(m0 + r) * CO + n] = acc[r];
}

// ---------------------------------------------------------------------------
// BatchNorm: per-channel sum/sumsq reduction then normalize+affine+ReLU.
// ---------------------------------------------------------------------------
__global__ void zero_f32(float* p, int n) {
    int i = blockIdx.x * blockDim.x + threadIdx.x;
    if (i < n) p[i] = 0.f;
}

__global__ void bn_stats(const float* __restrict__ X, float* __restrict__ sum,
                         float* __restrict__ sumsq, int rows, int C) {
    int tid = threadIdx.x;
    if (C <= 256) {                       // C in {16,32,64,128,256}
        int groups = 256 / C;
        int c = tid % C;
        int g = tid / C;
        float s = 0.f, s2 = 0.f;
        for (int r = blockIdx.x * groups + g; r < rows; r += gridDim.x * groups) {
            float v = X[(size_t)r * C + c];
            s += v; s2 += v * v;
        }
        __shared__ float ls[256], ls2[256];
        ls[tid] = s; ls2[tid] = s2;
        __syncthreads();
        if (g == 0) {
            for (int gg = 1; gg < groups; ++gg) {
                s  += ls[gg * C + c];
                s2 += ls2[gg * C + c];
            }
            atomicAdd(&sum[c], s);
            atomicAdd(&sumsq[c], s2);
        }
    } else {                              // C in {512, 1024}
        int nj = C / 256;
        float s[4], s2[4];
#pragma unroll
        for (int j = 0; j < 4; ++j) { s[j] = 0.f; s2[j] = 0.f; }
        for (int r = blockIdx.x; r < rows; r += gridDim.x) {
            const float* p = X + (size_t)r * C + tid;
#pragma unroll
            for (int j = 0; j < 4; ++j)
                if (j < nj) { float v = p[j * 256]; s[j] += v; s2[j] += v * v; }
        }
#pragma unroll
        for (int j = 0; j < 4; ++j)
            if (j < nj) {
                atomicAdd(&sum[tid + j * 256], s[j]);
                atomicAdd(&sumsq[tid + j * 256], s2[j]);
            }
    }
}

__global__ void bn_apply_relu(float* __restrict__ X,
                              const float* __restrict__ sum,
                              const float* __restrict__ sumsq,
                              const float* __restrict__ g,
                              const float* __restrict__ b,
                              long n, int C, float invCnt) {
    long i = (long)blockIdx.x * blockDim.x + threadIdx.x;
    long stride = (long)gridDim.x * blockDim.x;
    for (; i < n; i += stride) {
        int c = (int)(i % C);
        float m   = sum[c] * invCnt;
        float var = sumsq[c] * invCnt - m * m;
        float yv  = (X[i] - m) * rsqrtf(var + 1e-5f) * g[c] + b[c];
        X[i] = fmaxf(yv, 0.f);
    }
}

// ---------------------------------------------------------------------------
// Final: out[b, 2h+p, w] = relu(sum_c X[b,h,w,c]*fw[p,c] + fb)
// ---------------------------------------------------------------------------
__global__ void final_proj(const float* __restrict__ X,
                           const float* __restrict__ fw,
                           const float* __restrict__ fb,
                           float* __restrict__ out) {
    int i = blockIdx.x * blockDim.x + threadIdx.x;   // (b*128+h)*128+w
    const int total = 64 * 128 * 128;
    if (i >= total) return;
    const float* p = X + (size_t)i * 16;
    float d0 = fb[0], d1 = fb[0];
#pragma unroll
    for (int c = 0; c < 16; ++c) {
        float v = p[c];
        d0 += v * fw[c];
        d1 += v * fw[16 + c];
    }
    int b   = i >> 14;
    int rem = i & 16383;
    int h   = rem >> 7;
    int w   = rem & 127;
    size_t o0 = ((size_t)b * 256 + 2 * h) * 128 + w;
    out[o0]       = fmaxf(d0, 0.f);
    out[o0 + 128] = fmaxf(d1, 0.f);
}

// ---------------------------------------------------------------------------
// Host side
// ---------------------------------------------------------------------------
struct StageArgs {
    const float* up_w; const float* conv_w;
    const float* bn1g; const float* bn1b;
    const float* bn2g; const float* bn2b;
};

template<int CI, int CO>
static void run_stage(float*& cur, float*& oth, float* stats, const StageArgs& sa,
                      int lh, hipStream_t stream) {
    int h    = 1 << lh;
    int Min  = 64 * h * h;
    int H2   = 2 * h;
    int rows = 64 * H2 * H2;
    long n1  = (long)rows * CO;
    float invCnt = 1.f / (float)rows;

    // up2: cur -> oth (raw, pre-BN)
    up2_wmma<CI, CO><<<dim3(Min / 16, CO / 16), 32, 0, stream>>>(cur, sa.up_w, oth, lh);

    // BN1 + ReLU in place on oth
    zero_f32<<<(2 * CO + 255) / 256, 256, 0, stream>>>(stats, 2 * CO);
    bn_stats<<<256, 256, 0, stream>>>(oth, stats, stats + CO, rows, CO);
    bn_apply_relu<<<(unsigned)((n1 + 255) / 256), 256, 0, stream>>>(
        oth, stats, stats + CO, sa.bn1g, sa.bn1b, n1, CO, invCnt);

    // conv3x3: oth -> cur (raw, pre-BN)
    conv3_wmma<CO, CO><<<dim3(rows / 16, CO / 16), 32, 0, stream>>>(oth, sa.conv_w, cur, H2, H2);

    // BN2 + ReLU in place on cur
    zero_f32<<<(2 * CO + 255) / 256, 256, 0, stream>>>(stats, 2 * CO);
    bn_stats<<<256, 256, 0, stream>>>(cur, stats, stats + CO, rows, CO);
    bn_apply_relu<<<(unsigned)((n1 + 255) / 256), 256, 0, stream>>>(
        cur, stats, stats + CO, sa.bn2g, sa.bn2b, n1, CO, invCnt);
}

extern "C" void kernel_launch(void* const* d_in, const int* in_sizes, int n_in,
                              void* d_out, int out_size, void* d_ws, size_t ws_size,
                              hipStream_t stream) {
    const float* z       = (const float*)d_in[0];
    const float* lin1_w  = (const float*)d_in[1];
    const float* lin1_b  = (const float*)d_in[2];
    const float* lin2_w  = (const float*)d_in[3];
    const float* lin2_b  = (const float*)d_in[4];
    StageArgs sa[7];
    for (int i = 0; i < 7; ++i) {
        sa[i].up_w   = (const float*)d_in[5 + i];
        sa[i].conv_w = (const float*)d_in[12 + i];
        sa[i].bn1g   = (const float*)d_in[19 + i];
        sa[i].bn1b   = (const float*)d_in[26 + i];
        sa[i].bn2g   = (const float*)d_in[33 + i];
        sa[i].bn2b   = (const float*)d_in[40 + i];
    }
    const float* final_w = (const float*)d_in[47];
    const float* final_b = (const float*)d_in[48];

    const size_t BUF = 16777216;          // 64*128*128*16 floats (max activation)
    float* buf0  = (float*)d_ws;
    float* buf1  = buf0 + BUF;
    float* stats = buf1 + BUF;

    // FC1: (64,128)x(128,256)   FC2: (64,256)x(256,2048)
    gemm_bias_relu<128, 256> <<<dim3(4, 16),  32, 0, stream>>>(z,    lin1_w, lin1_b, buf0);
    gemm_bias_relu<256, 2048><<<dim3(4, 128), 32, 0, stream>>>(buf0, lin2_w, lin2_b, buf1);

    float* cur = buf1;   // (64,1,1,2048)
    float* oth = buf0;

    run_stage<2048, 1024>(cur, oth, stats, sa[0], 0, stream);
    run_stage<1024,  512>(cur, oth, stats, sa[1], 1, stream);
    run_stage< 512,  256>(cur, oth, stats, sa[2], 2, stream);
    run_stage< 256,  128>(cur, oth, stats, sa[3], 3, stream);
    run_stage< 128,   64>(cur, oth, stats, sa[4], 4, stream);
    run_stage<  64,   32>(cur, oth, stats, sa[5], 5, stream);
    run_stage<  32,   16>(cur, oth, stats, sa[6], 6, stream);

    // cur = (64,128,128,16) -> d_out (64,256,128,1)
    final_proj<<<(64 * 128 * 128 + 255) / 256, 256, 0, stream>>>(
        cur, final_w, final_b, (float*)d_out);
}
